// TopicPropagationModel_14894946582643
// MI455X (gfx1250) — matrix operand order repaired and checked by reading, not compiled, not run on hardware
//
#include <hip/hip_runtime.h>
#include <hip/hip_bf16.h>

// ---------------------------------------------------------------------------
// TopicPropagationModel on gfx1250 (MI455X): bf16 WMMA GEMMs + flash-style
// fused GAT attention. h is produced TRANSPOSED per head so the attention
// B-fragments are contiguous in LDS (vector ds_load_b128 gathers), and the
// H tiles are staged memory->LDS with double-buffered
// GLOBAL_LOAD_ASYNC_TO_LDS_B128 hidden behind the online-softmax VALU work.
// ---------------------------------------------------------------------------

typedef __attribute__((ext_vector_type(16))) __bf16 v16bf;
typedef __attribute__((ext_vector_type(8)))  __bf16 v8bf;
typedef __attribute__((ext_vector_type(8)))  float  v8f;

#define N_TOPICS 1024
#define TOPIC_DIM 128
#define HIDDEN 512
#define N_HEADS 8
#define HEAD_DIM 64
#define BATCH 8
#define NEGBIG (-1.0e30f)

// 16-bit A-matrix 16x32 fragment K mapping (matches B, contiguous in K):
// VGPR j: K=2j,2j+1 (lanes0-15) / K=16+2j,16+2j+1 (lanes16-31)
__device__ __forceinline__ int a_kbase(int j8, int khalf) {
    return 2 * j8 + 16 * khalf;
}
// 16-bit B-matrix 32x16 fragment K mapping:
// VGPR j: K=2j,2j+1 (lanes0-15) / K=16+2j,16+2j+1 (lanes16-31)
// => per-lane fragment = B[16*khalf + 0..15][col], contiguous in K.
__device__ __forceinline__ int b_kbase(int j8, int khalf) {
    return 2 * j8 + 16 * khalf;
}

// ---------------------------------------------------------------------------
// Generic bf16 WMMA GEMM:  C[b] = A[b] (f32, MxK) * B (KxN) (+ bias)
//   BMODE 0: B[k][n] = Bsrc[n*K + k]
//   BMODE 1: B[k][n] = Bsrc[(n/64)*K*64 + k*64 + (n%64)]   (gat_W[l] = (8, K, 64))
//   OUTMODE 0: f32 row-major + bias
//   OUTMODE 1: bf16 transposed-per-head:  C^T[(bz*8 + n/64)*64 + n%64][m]
//              (one contiguous v8bf store per C-register column)
// Block: 256 thr = 8 waves (4 in M x 2 in N); block tile 64x64; K step 32.
// ---------------------------------------------------------------------------
template <int BMODE, int OUTMODE>
__global__ __launch_bounds__(256) void gemm_bf16_wmma(
    const float* __restrict__ A, int lda, long aBatch,
    const float* __restrict__ Bsrc,
    const float* __restrict__ bias,
    void* __restrict__ Cout, int ldc, long cBatch,
    int M, int N, int K)
{
    __shared__ __align__(16) __bf16 As[64][40];
    __shared__ __align__(16) __bf16 Bs[32][72];

    const int tid  = threadIdx.x;
    const int lane = tid & 31;
    const int wid  = tid >> 5;     // 0..7
    const int wm   = wid & 3;      // 4 waves along M
    const int wn   = wid >> 2;     // 2 waves along N
    const int bm   = blockIdx.y * 64;
    const int bn   = blockIdx.x * 64;
    const int bz   = blockIdx.z;

    const float* Ab = A + (size_t)bz * (size_t)aBatch;
    const int row   = lane & 15;
    const int khalf = lane >> 4;

    v8f c0 = {}; v8f c1 = {};

    for (int k0 = 0; k0 < K; k0 += 32) {
        // cooperative A slab load (64 x 32), f32 -> bf16
        {
            const int r  = tid >> 2;          // 0..63
            const int kk = (tid & 3) * 8;     // 0,8,16,24
            const float* src = Ab + (size_t)(bm + r) * lda + (k0 + kk);
#pragma unroll
            for (int e = 0; e < 8; ++e) As[r][kk + e] = (__bf16)src[e];
        }
        // cooperative B slab load (32 x 64), gather per BMODE
        {
            const int kk = tid >> 3;          // 0..31
            const int n8 = (tid & 7) * 8;
#pragma unroll
            for (int e = 0; e < 8; ++e) {
                const int n = bn + n8 + e;
                float v;
                if (BMODE == 0)
                    v = Bsrc[(size_t)n * K + (k0 + kk)];
                else
                    v = Bsrc[(size_t)(n >> 6) * (size_t)K * 64 +
                             (size_t)(k0 + kk) * 64 + (n & 63)];
                Bs[kk][n8 + e] = (__bf16)v;
            }
        }
        __syncthreads();

        // A fragment (16x32) from LDS in contiguous-K lane order (same as B)
        v16bf af;
#pragma unroll
        for (int j = 0; j < 8; ++j) {
            const int kb = a_kbase(j, khalf);
            af[2 * j]     = As[wm * 16 + row][kb];
            af[2 * j + 1] = As[wm * 16 + row][kb + 1];
        }
        // two B fragments (32x16)
        v16bf bf0, bf1;
#pragma unroll
        for (int j = 0; j < 8; ++j) {
            const int kb = b_kbase(j, khalf);
            bf0[2 * j]     = Bs[kb][wn * 32 + row];
            bf0[2 * j + 1] = Bs[kb + 1][wn * 32 + row];
            bf1[2 * j]     = Bs[kb][wn * 32 + 16 + row];
            bf1[2 * j + 1] = Bs[kb + 1][wn * 32 + 16 + row];
        }
        c0 = __builtin_amdgcn_wmma_f32_16x16x32_bf16(false, af, false, bf0,
                                                     (short)0, c0, false, false);
        c1 = __builtin_amdgcn_wmma_f32_16x16x32_bf16(false, af, false, bf1,
                                                     (short)0, c1, false, false);
        __syncthreads();
    }

    // epilogue (C layout: lanes0-15 col=lane, M=r; lanes16-31 col=lane-16, M=8+r)
    const int col0 = bn + wn * 32 + row;
    const int col1 = col0 + 16;
    if (OUTMODE == 0) {
        const float b0 = bias ? bias[col0] : 0.f;
        const float b1 = bias ? bias[col1] : 0.f;
#pragma unroll
        for (int r = 0; r < 8; ++r) {
            const int mrow = bm + wm * 16 + ((lane < 16) ? r : r + 8);
            float* C = (float*)Cout + (size_t)bz * (size_t)cBatch;
            C[(size_t)mrow * ldc + col0] = c0[r] + b0;
            C[(size_t)mrow * ldc + col1] = c1[r] + b1;
        }
    } else {
        // transposed-per-head bf16: contiguous 8 x bf16 along M per lane
        const int mbase = bm + wm * 16 + ((lane < 16) ? 0 : 8);
        __bf16* C = (__bf16*)Cout + (size_t)bz * (size_t)cBatch;
        v8bf o0, o1;
#pragma unroll
        for (int r = 0; r < 8; ++r) { o0[r] = (__bf16)c0[r]; o1[r] = (__bf16)c1[r]; }
        __bf16* d0 = C + ((size_t)(col0 >> 6) * 64 + (col0 & 63)) * N_TOPICS + mbase;
        __bf16* d1 = C + ((size_t)(col1 >> 6) * 64 + (col1 & 63)) * N_TOPICS + mbase;
        *(v8bf*)d0 = o0;
        *(v8bf*)d1 = o1;
    }
}

// ---------------------------------------------------------------------------
// Global max of time_matrix (2 stages), then time_w = exp(-0.1*(maxT - T))
// ---------------------------------------------------------------------------
__global__ __launch_bounds__(256) void reduce_max_stage1(
    const float* __restrict__ x, long n, float* __restrict__ partials)
{
    __shared__ float sm[256];
    long i = (long)blockIdx.x * blockDim.x + threadIdx.x;
    const long stride = (long)gridDim.x * blockDim.x;
    float m = -3.4e38f;
    for (; i < n; i += stride) m = fmaxf(m, x[i]);
    sm[threadIdx.x] = m;
    __syncthreads();
    for (int off = 128; off > 0; off >>= 1) {
        if ((int)threadIdx.x < off)
            sm[threadIdx.x] = fmaxf(sm[threadIdx.x], sm[threadIdx.x + off]);
        __syncthreads();
    }
    if (threadIdx.x == 0) partials[blockIdx.x] = sm[0];
}

__global__ __launch_bounds__(256) void reduce_max_stage2(
    const float* __restrict__ partials, int n, float* __restrict__ mx)
{
    __shared__ float sm[256];
    float m = -3.4e38f;
    for (int i = threadIdx.x; i < n; i += 256) m = fmaxf(m, partials[i]);
    sm[threadIdx.x] = m;
    __syncthreads();
    for (int off = 128; off > 0; off >>= 1) {
        if ((int)threadIdx.x < off)
            sm[threadIdx.x] = fmaxf(sm[threadIdx.x], sm[threadIdx.x + off]);
        __syncthreads();
    }
    if (threadIdx.x == 0) mx[0] = sm[0];
}

__global__ __launch_bounds__(256) void compute_timew(
    const float* __restrict__ t, const float* __restrict__ mx,
    float* __restrict__ tw, long n)
{
    const long i = (long)blockIdx.x * blockDim.x + threadIdx.x;
    if (i < n) tw[i] = expf(-0.1f * (mx[0] - t[i]));
}

// ---------------------------------------------------------------------------
// Edge logits from TRANSPOSED h:  hT[(b*8+hd)*64 + d][n]
// e_src/e_dst[b,hd,n] = sum_d hT[d][n] * a_src/a_dst[d]
// (consecutive threads n read consecutive addresses -> fully coalesced)
// ---------------------------------------------------------------------------
__global__ __launch_bounds__(256) void edge_logits(
    const __bf16* __restrict__ hT, const float* __restrict__ a_layer,
    float* __restrict__ esrc, float* __restrict__ edst)
{
    const int idx = blockIdx.x * 256 + threadIdx.x;   // 65536 total
    const int n  = idx & 1023;
    const int hd = (idx >> 10) & 7;
    const int b  = idx >> 13;
    const __bf16* hr = hT + (size_t)((b * N_HEADS + hd) * HEAD_DIM) * N_TOPICS + n;
    const float* as  = a_layer + hd * 2 * HEAD_DIM;
    const float* ad  = as + HEAD_DIM;
    float s = 0.f, d = 0.f;
#pragma unroll 4
    for (int q = 0; q < HEAD_DIM; ++q) {
        const float hv = (float)hr[(size_t)q * N_TOPICS];
        s += hv * as[q];
        d += hv * ad[q];
    }
    esrc[idx] = s;
    edst[idx] = d;
}

// ---------------------------------------------------------------------------
// Fused GAT attention, flash-style. One wave per (b, head, 16-row block).
// hT chunk (64 d-rows x 32 j) staged via async DMA into LDS rows padded to
// 80B (20 dwords -> conflict-free b128 fragment reads). B fragments are 32
// contiguous bytes per lane: two v8bf (ds_load_b128) loads each.
// ---------------------------------------------------------------------------
__global__ __launch_bounds__(32) void gat_attn_flash(
    const __bf16* __restrict__ hT, const float* __restrict__ esrc,
    const float* __restrict__ edst, const float* __restrict__ tw,
    const int* __restrict__ adj, float* __restrict__ nodeOut)
{
    const int lane  = threadIdx.x;
    const int row   = lane & 15;
    const int khalf = lane >> 4;
    const int i0 = blockIdx.x * 16;
    const int hd = blockIdx.y;
    const int b  = blockIdx.z;
    const int i  = i0 + row;

    const float* twRow  = tw  + ((size_t)b * N_TOPICS + i) * N_TOPICS;
    const int*   adjRow = adj + ((size_t)b * N_TOPICS + i) * N_TOPICS;
    const float* ed = edst + (size_t)(b * N_HEADS + hd) * N_TOPICS;
    const float  es = esrc[(size_t)(b * N_HEADS + hd) * N_TOPICS + i];
    // transposed head slab: 64 rows (d) x 1024 (j)
    const __bf16* hTb = hT + (size_t)((b * N_HEADS + hd) * HEAD_DIM) * N_TOPICS;
    const unsigned long long gBase = (unsigned long long)(uintptr_t)hTb;

    // double-buffered transposed H tile: 2 x 64 rows x 40 bf16 (80B rows)
    __shared__ __align__(16) __bf16 HsT[2][64][40];
    const unsigned ldsBase = (unsigned)(uintptr_t)(&HsT[0][0][0]);

    // one chunk = 64 rows x 64B = 8 wave-wide async b128 transfers (ASYNCcnt +8)
    // instruction q: lane l -> row d = 8q + (l>>2), 16B segment (l&3)
    const int dRow  = lane >> 2;            // 0..7 within the 8-row group
    const unsigned seg = (lane & 3) * 16u;  // byte segment within 64B row
    auto issue_chunk = [&](int jc, int buf) {
        const unsigned long long gRow0 = gBase + (unsigned long long)jc * 2ull + seg;
        const unsigned lRow0 = ldsBase + (unsigned)buf * (64u * 80u) + dRow * 80u + seg;
#pragma unroll
        for (int q = 0; q < 8; ++q) {
            const unsigned long long g = gRow0 + (unsigned long long)(8 * q + dRow) * (N_TOPICS * 2ull);
            const unsigned l = lRow0 + (unsigned)(8 * q) * 80u;
            asm volatile("global_load_async_to_lds_b128 %0, %1, off"
                         :: "v"(l), "v"(g)
                         : "memory");
        }
    };

    float m = NEGBIG, lsum = 0.f;
    v8f acc[4] = {v8f{}, v8f{}, v8f{}, v8f{}};

    issue_chunk(0, 0);                               // prologue: ASYNCcnt = 8

    for (int j0 = 0; j0 < N_TOPICS; j0 += 32) {
        const int buf = (j0 >> 5) & 1;
        const bool more = (j0 + 32) < N_TOPICS;
        if (more) issue_chunk(j0 + 32, buf ^ 1);     // overlap next chunk DMA
        if (more) {                                  // L2 prefetch hints
            __builtin_prefetch(twRow + j0 + 32, 0, 1);
            __builtin_prefetch(adjRow + j0 + 32, 0, 1);
        }

        // scores directly in the A-fragment lane order (pairs are 8B aligned)
        float s[16];
        float pmax = NEGBIG;
#pragma unroll
        for (int j8 = 0; j8 < 8; ++j8) {
            const int kb = a_kbase(j8, khalf);
            const int j  = j0 + kb;
            const float2 edp = *(const float2*)(ed + j);
            const float2 twp = *(const float2*)(twRow + j);
            const int2   adp = *(const int2*)(adjRow + j);
            float e0 = es + edp.x;
            float e1 = es + edp.y;
            e0 = ((e0 > 0.f) ? e0 : (0.2f * e0)) * twp.x;   // leaky_relu * tw
            e1 = ((e1 > 0.f) ? e1 : (0.2f * e1)) * twp.y;
            const float s0 = adp.x ? e0 : NEGBIG;
            const float s1 = adp.y ? e1 : NEGBIG;
            s[2 * j8]     = s0;
            s[2 * j8 + 1] = s1;
            pmax = fmaxf(pmax, fmaxf(s0, s1));
        }
        // row stats: combine the two half-wave K ranges of the same row
        const float rowmax = fmaxf(pmax, __shfl_xor(pmax, 16, 32));
        const float mnew   = fmaxf(m, rowmax);
        const float scale  = expf(m - mnew);   // m==mnew==NEGBIG -> exp(0)=1, benign

        v16bf pf;
        float psum = 0.f;
#pragma unroll
        for (int t = 0; t < 16; ++t) {
            const float p = expf(s[t] - mnew);
            psum += p;
            pf[t] = (__bf16)p;
        }
        const float rowsum = psum + __shfl_xor(psum, 16, 32);
        lsum = lsum * scale + rowsum;
        m = mnew;

        // rescale accumulators; per-row scale lives in lanes 0..15 (row==lane)
#pragma unroll
        for (int r = 0; r < 8; ++r) {
            const int mrow = (lane < 16) ? r : r + 8;
            const float sc = __shfl(scale, mrow, 32);
            acc[0][r] *= sc; acc[1][r] *= sc; acc[2][r] *= sc; acc[3][r] *= sc;
        }

        // wait for the current chunk's DMA: <=8 outstanding leaves only the
        // prefetched next chunk in flight; on the last chunk drain fully.
        if (more) asm volatile("s_wait_asynccnt 0x8" ::: "memory");
        else      asm volatile("s_wait_asynccnt 0x0" ::: "memory");

        // P @ H : 4 output tiles; per-lane fragment = HsT[col][16*khalf+0..15]
        // = 32 contiguous bytes -> two 16B LDS vector loads.
#pragma unroll
        for (int nt = 0; nt < 4; ++nt) {
            const __bf16* fp = &HsT[buf][nt * 16 + row][khalf * 16];
            const v8bf lo = *(const v8bf*)(fp);
            const v8bf hi = *(const v8bf*)(fp + 8);
            v16bf bf;
#pragma unroll
            for (int t = 0; t < 8; ++t) { bf[t] = lo[t]; bf[t + 8] = hi[t]; }
            acc[nt] = __builtin_amdgcn_wmma_f32_16x16x32_bf16(
                false, pf, false, bf, (short)0, acc[nt], false, false);
        }
    }

    // epilogue: normalize by row sum, ELU, scatter to node features
#pragma unroll
    for (int r = 0; r < 8; ++r) {
        const int mrow = (lane < 16) ? r : r + 8;
        const float li  = __shfl(lsum, mrow, 32);
        const float inv = 1.0f / li;
        float* outRow = nodeOut + ((size_t)b * N_TOPICS + (i0 + mrow)) * HIDDEN
                        + hd * HEAD_DIM;
#pragma unroll
        for (int nt = 0; nt < 4; ++nt) {
            float v = acc[nt][r] * inv;
            v = (v > 0.f) ? v : (expf(v) - 1.0f);      // ELU
            outRow[nt * 16 + row] = v;
        }
    }
}

// ---------------------------------------------------------------------------
// Final head: gather topic feature, + attract, fc1+relu, fc2. One block.
// ---------------------------------------------------------------------------
__global__ __launch_bounds__(512) void head_mlp(
    const float* __restrict__ node2, const int* __restrict__ topic_ids,
    const float* __restrict__ attractiveness,
    const float* __restrict__ attract_W, const float* __restrict__ attract_b,
    const float* __restrict__ fc1_W, const float* __restrict__ fc1_b,
    const float* __restrict__ fc2_W, const float* __restrict__ fc2_b,
    float* __restrict__ out)
{
    __shared__ float comb[HIDDEN];
    __shared__ float red[HIDDEN];
    const int t = threadIdx.x;
    for (int b = 0; b < BATCH; ++b) {
        const int tid_topic = topic_ids[b];
        comb[t] = node2[((size_t)b * N_TOPICS + tid_topic) * HIDDEN + t]
                + attractiveness[b] * attract_W[t] + attract_b[t];
        __syncthreads();
        float acc = fc1_b[t];
#pragma unroll 4
        for (int i = 0; i < HIDDEN; ++i) acc += comb[i] * fc1_W[(size_t)t * HIDDEN + i];
        red[t] = fmaxf(acc, 0.f) * fc2_W[t];
        __syncthreads();
        for (int off = 256; off > 0; off >>= 1) {
            if (t < off) red[t] += red[t + off];
            __syncthreads();
        }
        if (t == 0) out[b] = red[0] + fc2_b[0];
        __syncthreads();
    }
}

// ---------------------------------------------------------------------------
extern "C" void kernel_launch(void* const* d_in, const int* in_sizes, int n_in,
                              void* d_out, int out_size, void* d_ws, size_t ws_size,
                              hipStream_t stream)
{
    (void)in_sizes; (void)n_in; (void)out_size; (void)ws_size;

    const int*   topic_ids      = (const int*)  d_in[0];
    const int*   adj_matrix     = (const int*)  d_in[1];
    const float* time_matrix    = (const float*)d_in[2];
    const float* attractiveness = (const float*)d_in[3];
    const float* topic_emb      = (const float*)d_in[4];
    const float* struct_W       = (const float*)d_in[5];
    const float* struct_b       = (const float*)d_in[6];
    const float* attract_W      = (const float*)d_in[7];
    const float* attract_b      = (const float*)d_in[8];
    const float* gat_W          = (const float*)d_in[9];   // (2,8,512,64)
    const float* gat_a          = (const float*)d_in[10];  // (2,8,128,1)
    const float* fc1_W          = (const float*)d_in[11];
    const float* fc1_b          = (const float*)d_in[12];
    const float* fc2_W          = (const float*)d_in[13];
    const float* fc2_b          = (const float*)d_in[14];
    float* out = (float*)d_out;

    // workspace carve-up (256B aligned)
    char* ws = (char*)d_ws;
    size_t off = 0;
    auto carve = [&](size_t bytes) -> void* {
        void* p = ws + off;
        off = (off + bytes + 255) & ~(size_t)255;
        return p;
    };
    const long NN = (long)BATCH * N_TOPICS * N_TOPICS;              // 8M
    float*  tw       = (float*) carve((size_t)NN * 4);              // 32 MB
    float*  node0    = (float*) carve((size_t)N_TOPICS * HIDDEN * 4);            // 2 MB
    float*  node1    = (float*) carve((size_t)BATCH * N_TOPICS * HIDDEN * 4);    // 16 MB
    float*  node2    = (float*) carve((size_t)BATCH * N_TOPICS * HIDDEN * 4);    // 16 MB
    __bf16* hbufT    = (__bf16*)carve((size_t)BATCH * N_TOPICS * HIDDEN * 2);    // 8 MB
    float*  esrc     = (float*) carve((size_t)BATCH * N_HEADS * N_TOPICS * 4);
    float*  edst     = (float*) carve((size_t)BATCH * N_HEADS * N_TOPICS * 4);
    float*  partials = (float*) carve(256 * 4);
    float*  maxT     = (float*) carve(256);

    // 1) global max of time_matrix, then time_w (exp done ONCE, reused 16x)
    reduce_max_stage1<<<256, 256, 0, stream>>>(time_matrix, NN, partials);
    reduce_max_stage2<<<1, 256, 0, stream>>>(partials, 256, maxT);
    compute_timew<<<(int)((NN + 255) / 256), 256, 0, stream>>>(time_matrix, maxT, tw, NN);

    // 2) node0 = topic_embedding @ struct_W^T + struct_b   (1024x512, f32 out)
    gemm_bf16_wmma<0, 0><<<dim3(HIDDEN / 64, N_TOPICS / 64, 1), 256, 0, stream>>>(
        topic_emb, TOPIC_DIM, 0L, struct_W, struct_b,
        (void*)node0, HIDDEN, 0L, N_TOPICS, HIDDEN, TOPIC_DIM);

    const long nodeBatch = (long)N_TOPICS * HIDDEN;
    const float* nodeIn[2] = { node0, node1 };
    const long   aBatch[2] = { 0L, nodeBatch };
    float* nodeOutL[2] = { node1, node2 };

    for (int l = 0; l < 2; ++l) {
        // hT[b] = bf16(node[b] @ W_cat[l])^T per head  (batched, transposed store)
        gemm_bf16_wmma<1, 1><<<dim3(HIDDEN / 64, N_TOPICS / 64, BATCH), 256, 0, stream>>>(
            nodeIn[l], HIDDEN, aBatch[l],
            gat_W + (size_t)l * N_HEADS * HIDDEN * HEAD_DIM, nullptr,
            (void*)hbufT, HIDDEN, nodeBatch, N_TOPICS, HIDDEN, HIDDEN);

        // e_src / e_dst from transposed h
        edge_logits<<<(BATCH * N_HEADS * N_TOPICS) / 256, 256, 0, stream>>>(
            hbufT, gat_a + (size_t)l * N_HEADS * 2 * HEAD_DIM, esrc, edst);

        // fused attention + softmax + P@H + ELU
        gat_attn_flash<<<dim3(N_TOPICS / 16, N_HEADS, BATCH), 32, 0, stream>>>(
            hbufT, esrc, edst, tw, adj_matrix, nodeOutL[l]);
    }

    // 3) final MLP head
    head_mlp<<<1, 512, 0, stream>>>(node2, topic_ids, attractiveness,
                                    attract_W, attract_b, fc1_W, fc1_b,
                                    fc2_W, fc2_b, out);
}
